// GotenInteractionModule_27805618275234
// MI455X (gfx1250) — matrix operand order repaired and checked by reading, not compiled, not run on hardware
//
#include <hip/hip_runtime.h>
#include <hip/hip_bf16.h>
#include <math.h>

typedef __attribute__((ext_vector_type(16))) __bf16 bf16x16;
typedef __attribute__((ext_vector_type(8)))  __bf16 bf16x8;
typedef __attribute__((ext_vector_type(8)))  float  f32x8;

// ---------------------------------------------------------------------------
// WMMA helper: computes a 16x16 f32 tile of  A(16xKp,bf16,LDS) @ B(KpxNout,bf16)
// B is pre-swizzled in global memory so each lane reads 32 contiguous bytes per
// k-step.  A-fragment layout per CDNA5 ISA (16-bit A 16x32): lane<16 -> M=lane,
// k = {grp*16 + half*8 + j}.  B-fragment: N=lane%16, same k order as A:
// k = (i>>3)*16 + (lane>>4)*8 + (i&7).
// ---------------------------------------------------------------------------
__device__ __forceinline__ f32x8 wmma_rowtile(const __bf16* Alds, int Kp,
                                              const __bf16* __restrict__ Bswz,
                                              int ntile, int lane) {
  f32x8 c = {};
  const int m   = lane & 15;
  const int hlf = lane >> 4;
  const __bf16* ab = Alds + m * Kp + hlf * 8;
  const __bf16* bb = Bswz + (size_t)ntile * (size_t)(Kp * 16) + (size_t)(lane * 16);
  for (int kt = 0; kt < (Kp >> 5); ++kt) {
    bf16x8 alo = *(const bf16x8*)(ab + kt * 32);
    bf16x8 ahi = *(const bf16x8*)(ab + kt * 32 + 16);
    bf16x16 a;
#pragma unroll
    for (int i = 0; i < 8; ++i) { a[i] = alo[i]; a[i + 8] = ahi[i]; }
    bf16x16 b = *(const bf16x16*)(bb + kt * 512);
    c = __builtin_amdgcn_wmma_f32_16x16x32_bf16(false, a, false, b, (short)0, c,
                                                false, false);
  }
  return c;
}

// Dual-B variant: one A-fragment load feeds two WMMAs (for GEMM pairs sharing A)
__device__ __forceinline__ void wmma_rowtile2(const __bf16* Alds, int Kp,
                                              const __bf16* __restrict__ B0swz,
                                              const __bf16* __restrict__ B1swz,
                                              int ntile, int lane,
                                              f32x8& c0, f32x8& c1) {
  c0 = {}; c1 = {};
  const int m   = lane & 15;
  const int hlf = lane >> 4;
  const __bf16* ab  = Alds + m * Kp + hlf * 8;
  const size_t boff = (size_t)ntile * (size_t)(Kp * 16) + (size_t)(lane * 16);
  const __bf16* bb0 = B0swz + boff;
  const __bf16* bb1 = B1swz + boff;
  for (int kt = 0; kt < (Kp >> 5); ++kt) {
    bf16x8 alo = *(const bf16x8*)(ab + kt * 32);
    bf16x8 ahi = *(const bf16x8*)(ab + kt * 32 + 16);
    bf16x16 a;
#pragma unroll
    for (int i = 0; i < 8; ++i) { a[i] = alo[i]; a[i + 8] = ahi[i]; }
    bf16x16 b0 = *(const bf16x16*)(bb0 + kt * 512);
    bf16x16 b1 = *(const bf16x16*)(bb1 + kt * 512);
    c0 = __builtin_amdgcn_wmma_f32_16x16x32_bf16(false, a, false, b0, (short)0, c0,
                                                 false, false);
    c1 = __builtin_amdgcn_wmma_f32_16x16x32_bf16(false, a, false, b1, (short)0, c1,
                                                 false, false);
  }
}

__device__ __forceinline__ float silu_f(float x) { return x / (1.f + __expf(-x)); }

__device__ __forceinline__ void atomicMaxF(float* addr, float v) {
  if (v >= 0.f) atomicMax((int*)addr, __float_as_int(v));
  else          atomicMin((unsigned int*)addr, __float_as_uint(v));
}

// ---------------------------------------------------------------------------
__global__ void __launch_bounds__(256) k_fill(float* __restrict__ p, float v,
                                              size_t n) {
  size_t i = (size_t)blockIdx.x * 256 + threadIdx.x;
  if (i < n) p[i] = v;
}

// fp32 row-major W[K][Nout] -> bf16 swizzled fragment layout (K padded to Kp)
__global__ void __launch_bounds__(256) k_swizzle(const float* __restrict__ W,
                                                 __bf16* __restrict__ dst,
                                                 int K, int Kp, int Nout) {
  int idx = blockIdx.x * 256 + threadIdx.x;
  int total = Kp * Nout;
  if (idx >= total) return;
  int ntile = idx / (Kp * 16);
  int rem   = idx - ntile * (Kp * 16);
  int kt    = rem >> 9;
  int rem2  = rem & 511;
  int lane  = rem2 >> 4;
  int i     = rem2 & 15;
  int n = ntile * 16 + (lane & 15);
  int k = kt * 32 + (i >> 3) * 16 + (lane >> 4) * 8 + (i & 7);
  float v = (k < K) ? W[(size_t)k * Nout + n] : 0.f;
  dst[idx] = (__bf16)v;
}

// ---------------------------------------------------------------------------
// K1: per-node precompute  q=h@Wq+b, k=h@Wk+b, s=MLP_gs(h), v=MLP_gv(h)
// ---------------------------------------------------------------------------
__global__ void __launch_bounds__(256) k_node_pre(
    const float* __restrict__ h,
    const __bf16* __restrict__ swWq, const float* __restrict__ Wq_b,
    const __bf16* __restrict__ swWk, const float* __restrict__ Wk_b,
    const __bf16* __restrict__ swGs1, const float* __restrict__ gs_b1,
    const __bf16* __restrict__ swGs2, const float* __restrict__ gs_b2,
    const __bf16* __restrict__ swGv1, const float* __restrict__ gv_b1,
    const __bf16* __restrict__ swGv2, const float* __restrict__ gv_b2,
    float* __restrict__ qn, float* __restrict__ kn,
    float* __restrict__ sn, float* __restrict__ vn) {
  __shared__ __bf16 s_a[2048];
  __shared__ __bf16 s_a2[2048];
  __shared__ float  s_c[2048];
  const int tid = threadIdx.x, lane = tid & 31, wave = tid >> 5;
  const size_t n0 = (size_t)blockIdx.x * 16;
  for (int i = tid; i < 2048; i += 256) s_a[i] = (__bf16)h[n0 * 128 + i];
  __syncthreads();
  const int col = wave * 16 + (lane & 15);
  const int mb  = (lane >> 4) * 8;
  {  // q / k share A
    const float bq = Wq_b[col], bk = Wk_b[col];
    f32x8 cq, ck;
    wmma_rowtile2(s_a, 128, swWq, swWk, wave, lane, cq, ck);
#pragma unroll
    for (int v = 0; v < 8; ++v) {
      qn[(n0 + mb + v) * 128 + col] = cq[v] + bq;
      kn[(n0 + mb + v) * 128 + col] = ck[v] + bk;
    }
  }
  // gs layer1 (gv layer1 re-done later to keep one staging buffer live at a time)
  {
    const float b1 = gs_b1[col];
    f32x8 c = wmma_rowtile(s_a, 128, swGs1, wave, lane);
#pragma unroll
    for (int v = 0; v < 8; ++v) s_c[(mb + v) * 128 + col] = silu_f(c[v] + b1);
  }
  __syncthreads();
  for (int i = tid; i < 2048; i += 256) s_a2[i] = (__bf16)s_c[i];
  __syncthreads();
  {
    const float b2 = gs_b2[col];
    f32x8 c = wmma_rowtile(s_a2, 128, swGs2, wave, lane);
#pragma unroll
    for (int v = 0; v < 8; ++v) sn[(n0 + mb + v) * 128 + col] = c[v] + b2;
  }
  {
    const float b1 = gv_b1[col];
    f32x8 c = wmma_rowtile(s_a, 128, swGv1, wave, lane);
#pragma unroll
    for (int v = 0; v < 8; ++v) s_c[(mb + v) * 128 + col] = silu_f(c[v] + b1);
  }
  __syncthreads();
  for (int i = tid; i < 2048; i += 256) s_a2[i] = (__bf16)s_c[i];
  __syncthreads();
  {
    const float b2 = gv_b2[col];
    f32x8 c = wmma_rowtile(s_a2, 128, swGv2, wave, lane);
#pragma unroll
    for (int v = 0; v < 8; ++v) vn[(n0 + mb + v) * 128 + col] = c[v] + b2;
  }
}

// ---------------------------------------------------------------------------
// K2: edge phase 1 — eq_lin / vec_rej / w ; gw & gt MLPs ; t residual ; Wre+silu
//     ; alpha + segment max.  16 edges per block.
// ---------------------------------------------------------------------------
__global__ void __launch_bounds__(256) k_edge1(
    const float* __restrict__ Xin, const float* __restrict__ t_in,
    const float* __restrict__ sph,
    const float* __restrict__ Wtq, const float* __restrict__ Wtk,
    const int* __restrict__ ec, const int* __restrict__ en,
    const __bf16* __restrict__ swGw1, const float* __restrict__ gw_b1,
    const __bf16* __restrict__ swGw2, const float* __restrict__ gw_b2,
    const __bf16* __restrict__ swGt1, const float* __restrict__ gt_b1,
    const __bf16* __restrict__ swGt2, const float* __restrict__ gt_b2,
    const __bf16* __restrict__ swWre,
    const float* __restrict__ qn, const float* __restrict__ kn,
    const float* __restrict__ uc_p,
    float* __restrict__ t_out, float* __restrict__ alpha,
    float* __restrict__ amax) {
  __shared__ float s_xc[2304], s_xn[2304], s_xq[2304], s_xk[2304];
  __shared__ float s_d[144], s_w[768], s_t[2048], s_wt[1536];
  __shared__ int   s_ec[16], s_en[16];
  float*  s_y  = s_xq;             // y_gw   (2048 <= 2304)
  float*  s_c  = s_xk;             // staging
  __bf16* s_ab = (__bf16*)s_xn;    // bf16 A buffer (4KB <= 9216B)

  const int tid = threadIdx.x, lane = tid & 31, wave = tid >> 5;
  const size_t e0 = (size_t)blockIdx.x * 16;
  const float uc = uc_p[0];
  const float co = rsqrtf(uc * uc + 1.f);
  const float ucco = uc * co;
  const int col = wave * 16 + (lane & 15);
  const int mb  = (lane >> 4) * 8;

  if (tid < 16) { s_ec[tid] = ec[e0 + tid]; s_en[tid] = en[e0 + tid]; }
  for (int i = tid; i < 1536; i += 256) s_wt[i] = (i < 768) ? Wtq[i] : Wtk[i - 768];
  for (int i = tid; i < 144; i += 256)  s_d[i] = sph[e0 * 9 + i];
  __syncthreads();
  for (int i = tid; i < 2304; i += 256) {
    int e = i / 144, r = i - e * 144;
    s_xc[i] = Xin[(size_t)s_ec[e] * 144 + r];
    s_xn[i] = Xin[(size_t)s_en[e] * 144 + r];
  }
  for (int i = tid; i < 2048; i += 256) s_t[i] = t_in[e0 * 128 + i];
  __syncthreads();
  // eq_lin (X@Wt / sqrt(M)) per SO3 block
  for (int idx = tid; idx < 2304; idx += 256) {
    int e = idx / 144, r = idx - e * 144;
    int k = r / 9, i = r - k * 9;
    int l = (i == 0) ? 0 : (i < 4 ? 1 : 2);
    float aq = 0.f, ak = 0.f;
    for (int m = 0; m < 16; ++m) {
      aq += s_xc[e * 144 + m * 9 + i] * s_wt[l * 256 + m * 16 + k];
      ak += s_xn[e * 144 + m * 9 + i] * s_wt[768 + l * 256 + m * 16 + k];
    }
    s_xq[idx] = aq * 0.25f;
    s_xk[idx] = ak * 0.25f;
  }
  __syncthreads();
  // vec_rej (rej w.r.t. -d equals rej w.r.t. d) + per-block inner products
  {
    int e = tid >> 4, k = tid & 15;
    float dq = 0.f, dk = 0.f;
    for (int i = 0; i < 9; ++i) {
      dq += s_xq[e * 144 + k * 9 + i] * s_d[e * 9 + i];
      dk += s_xk[e * 144 + k * 9 + i] * s_d[e * 9 + i];
    }
    float w0 = 0.f, w1 = 0.f, w2 = 0.f;
    for (int i = 0; i < 9; ++i) {
      float fq = s_xq[e * 144 + k * 9 + i] - dq * s_d[e * 9 + i];
      float fk = s_xk[e * 144 + k * 9 + i] - dk * s_d[e * 9 + i];
      float p = fq * fk;
      if (i == 0) w0 += p; else if (i < 4) w1 += p; else w2 += p;
    }
    s_w[e * 48 + k] = w0; s_w[e * 48 + 16 + k] = w1; s_w[e * 48 + 32 + k] = w2;
  }
  __syncthreads();
  // A = bf16(w) padded to K=64
  for (int idx = tid; idx < 16 * 64; idx += 256) {
    int e = idx >> 6, k = idx & 63;
    s_ab[idx] = (__bf16)((k < 48) ? s_w[e * 48 + k] : 0.f);
  }
  __syncthreads();
  {  // u1 = silu(w@gwW1+b1)
    const float b1 = gw_b1[col];
    f32x8 c = wmma_rowtile(s_ab, 64, swGw1, wave, lane);
#pragma unroll
    for (int v = 0; v < 8; ++v) s_c[(mb + v) * 128 + col] = silu_f(c[v] + b1);
  }
  __syncthreads();
  for (int i = tid; i < 2048; i += 256) s_ab[i] = (__bf16)s_c[i];
  __syncthreads();
  {  // y_gw = u1@gwW2+b2
    const float b2 = gw_b2[col];
    f32x8 c = wmma_rowtile(s_ab, 128, swGw2, wave, lane);
#pragma unroll
    for (int v = 0; v < 8; ++v) s_y[(mb + v) * 128 + col] = c[v] + b2;
  }
  __syncthreads();
  for (int i = tid; i < 2048; i += 256) s_ab[i] = (__bf16)s_t[i];
  __syncthreads();
  {  // u2 = silu(t@gtW1+b1)
    const float b1 = gt_b1[col];
    f32x8 c = wmma_rowtile(s_ab, 128, swGt1, wave, lane);
#pragma unroll
    for (int v = 0; v < 8; ++v) s_c[(mb + v) * 128 + col] = silu_f(c[v] + b1);
  }
  __syncthreads();
  for (int i = tid; i < 2048; i += 256) s_ab[i] = (__bf16)s_c[i];
  __syncthreads();
  {  // y_gt ; delta_t = y_gw*y_gt ; t_new residual -> s_t & global out
    const float b2 = gt_b2[col];
    f32x8 c = wmma_rowtile(s_ab, 128, swGt2, wave, lane);
#pragma unroll
    for (int v = 0; v < 8; ++v) {
      int off = (mb + v) * 128 + col;
      float tn = co * s_t[off] + ucco * (s_y[off] * (c[v] + b2));
      s_t[off] = tn;
      t_out[e0 * 128 + off] = tn;
    }
  }
  __syncthreads();
  for (int i = tid; i < 2048; i += 256) s_ab[i] = (__bf16)s_t[i];
  __syncthreads();
  {  // ta = silu(t_new@Wre)
    f32x8 c = wmma_rowtile(s_ab, 128, swWre, wave, lane);
#pragma unroll
    for (int v = 0; v < 8; ++v) s_c[(mb + v) * 128 + col] = silu_f(c[v]);
  }
  __syncthreads();
  if (tid < 64) {  // alpha per (edge, head) + segment max
    int e = tid >> 2, hh = tid & 3;
    int ci = s_ec[e], ni = s_en[e];
    float s = 0.f;
    for (int d_ = 0; d_ < 32; ++d_) {
      int cc = hh * 32 + d_;
      s += qn[(size_t)ci * 128 + cc] * kn[(size_t)ni * 128 + cc] * s_c[e * 128 + cc];
    }
    float a = s * 0.17677669529663687f;  // 1/sqrt(32)
    alpha[(e0 + e) * 4 + hh] = a;
    atomicMaxF(&amax[(size_t)ci * 4 + hh], a);
  }
}

// ---------------------------------------------------------------------------
// K3: ex = exp(alpha - amax[c]);  den += ex   (alpha buffer updated in place)
// ---------------------------------------------------------------------------
__global__ void __launch_bounds__(256) k_softmax(const int* __restrict__ ec,
                                                 float* __restrict__ alpha,
                                                 const float* __restrict__ amax,
                                                 float* __restrict__ den, int E4) {
  int idx = blockIdx.x * 256 + threadIdx.x;
  if (idx >= E4) return;
  int e = idx >> 2, hh = idx & 3;
  int c = ec[e];
  float ex = __expf(alpha[idx] - amax[(size_t)c * 4 + hh]);
  alpha[idx] = ex;
  atomicAdd(&den[(size_t)c * 4 + hh], ex);
}

// ---------------------------------------------------------------------------
// K5: edge phase 2 — spatial (recomputed), sea, gsp gates, md/mt MLPs,
//     atomic segment-sums for dh and dX.
// ---------------------------------------------------------------------------
__global__ void __launch_bounds__(256) k_edge2(
    const float* __restrict__ t_new, const float* __restrict__ Xin,
    const float* __restrict__ sph,
    const int* __restrict__ ec, const int* __restrict__ en,
    const float* __restrict__ exb, const float* __restrict__ den,
    const float* __restrict__ sn, const float* __restrict__ vn,
    const __bf16* __restrict__ swWrs, const float* __restrict__ Wrs_b,
    const __bf16* __restrict__ swSp1, const float* __restrict__ gsp_b1,
    const __bf16* __restrict__ swSp2, const float* __restrict__ gsp_b2,
    const __bf16* __restrict__ swMd1, const float* __restrict__ md_b1,
    const __bf16* __restrict__ swMd2, const float* __restrict__ md_b2,
    const __bf16* __restrict__ swMt1, const float* __restrict__ mt_b1,
    const __bf16* __restrict__ swMt2, const float* __restrict__ mt_b2,
    float* __restrict__ dh_acc, float* __restrict__ dX_acc) {
  __shared__ float s_t[2048];   // t_new, then "combined"
  __shared__ float s_c[2048];
  __shared__ float s_g[6144];   // gates 16x384
  __shared__ float s_xn[2304];
  __shared__ float s_d[144];
  __shared__ float s_gd[256], s_gt[256], s_attn[64];
  __shared__ __bf16 s_ab[2048];
  __shared__ int s_ec[16], s_en[16];
  const int tid = threadIdx.x, lane = tid & 31, wave = tid >> 5;
  const size_t e0 = (size_t)blockIdx.x * 16;
  const int col = wave * 16 + (lane & 15);
  const int mb  = (lane >> 4) * 8;

  if (tid < 16) { s_ec[tid] = ec[e0 + tid]; s_en[tid] = en[e0 + tid]; }
  for (int i = tid; i < 144; i += 256) s_d[i] = sph[e0 * 9 + i];
  for (int i = tid; i < 2048; i += 256) s_t[i] = t_new[e0 * 128 + i];
  __syncthreads();
  for (int i = tid; i < 2304; i += 256) {
    int e = i / 144, r = i - e * 144;
    s_xn[i] = Xin[(size_t)s_en[e] * 144 + r];
  }
  if (tid < 64) {
    int e = tid >> 2, hh = tid & 3;
    s_attn[tid] = exb[(e0 + e) * 4 + hh] /
                  (den[(size_t)s_ec[e] * 4 + hh] + 1e-16f);
  }
  __syncthreads();
  for (int i = tid; i < 2048; i += 256) s_ab[i] = (__bf16)s_t[i];
  __syncthreads();
  {  // combined = (t_new@Wrs+b)*s_n[j] + attn*v_n[j]
    const float br = Wrs_b[col];
    f32x8 c = wmma_rowtile(s_ab, 128, swWrs, wave, lane);
    int hh = col >> 5;
#pragma unroll
    for (int v = 0; v < 8; ++v) {
      int e = mb + v;
      size_t nj = (size_t)s_en[e];
      float spat = (c[v] + br) * sn[nj * 128 + col];
      float sea  = s_attn[e * 4 + hh] * vn[nj * 128 + col];
      s_t[e * 128 + col] = spat + sea;
    }
  }
  __syncthreads();
  for (int i = tid; i < 2048; i += 256) s_ab[i] = (__bf16)s_t[i];
  __syncthreads();
  {  // gsp layer 1
    const float b1 = gsp_b1[col];
    f32x8 c = wmma_rowtile(s_ab, 128, swSp1, wave, lane);
#pragma unroll
    for (int v = 0; v < 8; ++v) s_c[(mb + v) * 128 + col] = silu_f(c[v] + b1);
  }
  __syncthreads();
  for (int i = tid; i < 2048; i += 256) s_ab[i] = (__bf16)s_c[i];
  __syncthreads();
  for (int nt = wave; nt < 24; nt += 8) {  // gsp layer 2, 384 cols
    int cl = nt * 16 + (lane & 15);
    const float b2 = gsp_b2[cl];
    f32x8 c = wmma_rowtile(s_ab, 128, swSp2, nt, lane);
#pragma unroll
    for (int v = 0; v < 8; ++v) s_g[(mb + v) * 384 + cl] = c[v] + b2;
  }
  __syncthreads();
  // o_s scatter + A = bf16(o_d)
  for (int i = tid; i < 2048; i += 256) {
    int e = i >> 7, cl = i & 127;
    atomicAdd(&dh_acc[(size_t)s_ec[e] * 128 + cl], s_g[e * 384 + cl]);
    s_ab[i] = (__bf16)s_g[e * 384 + 128 + cl];
  }
  __syncthreads();
  {  // md layer 1
    const float b1 = md_b1[col];
    f32x8 c = wmma_rowtile(s_ab, 128, swMd1, wave, lane);
#pragma unroll
    for (int v = 0; v < 8; ++v) s_c[(mb + v) * 128 + col] = silu_f(c[v] + b1);
  }
  __syncthreads();
  for (int i = tid; i < 2048; i += 256) s_ab[i] = (__bf16)s_c[i];
  __syncthreads();
  if (wave == 0) {  // md layer 2 (dout = 16)
    int cl = lane & 15;
    const float b2 = md_b2[cl];
    f32x8 c = wmma_rowtile(s_ab, 128, swMd2, 0, lane);
#pragma unroll
    for (int v = 0; v < 8; ++v) s_gd[(mb + v) * 16 + cl] = c[v] + b2;
  }
  __syncthreads();
  for (int i = tid; i < 2048; i += 256)
    s_ab[i] = (__bf16)s_g[(i >> 7) * 384 + 256 + (i & 127)];
  __syncthreads();
  {  // mt layer 1
    const float b1 = mt_b1[col];
    f32x8 c = wmma_rowtile(s_ab, 128, swMt1, wave, lane);
#pragma unroll
    for (int v = 0; v < 8; ++v) s_c[(mb + v) * 128 + col] = silu_f(c[v] + b1);
  }
  __syncthreads();
  for (int i = tid; i < 2048; i += 256) s_ab[i] = (__bf16)s_c[i];
  __syncthreads();
  if (wave == 0) {  // mt layer 2 (dout = 16)
    int cl = lane & 15;
    const float b2 = mt_b2[cl];
    f32x8 c = wmma_rowtile(s_ab, 128, swMt2, 0, lane);
#pragma unroll
    for (int v = 0; v < 8; ++v) s_gt[(mb + v) * 16 + cl] = c[v] + b2;
  }
  __syncthreads();
  {  // dX_ij scatter
    int e = tid >> 4, k = tid & 15;
    size_t base = (size_t)s_ec[e] * 144 + k * 9;
    float gd = s_gd[e * 16 + k], gt = s_gt[e * 16 + k];
    for (int i = 0; i < 9; ++i)
      atomicAdd(&dX_acc[base + i], gd * s_d[e * 9 + i] + gt * s_xn[e * 144 + k * 9 + i]);
  }
}

// ---------------------------------------------------------------------------
// K6: node finalize — LN, residual h; SO3-LN, residual X; Xp, nrm, gm MLP;
//     final h/X outputs.
// ---------------------------------------------------------------------------
__global__ void __launch_bounds__(256) k_node_fin(
    const float* __restrict__ h, const float* __restrict__ Xin,
    const float* __restrict__ dh_acc, const float* __restrict__ dX_acc,
    const float* __restrict__ ln_g, const float* __restrict__ ln_b,
    const float* __restrict__ Wvu,
    const __bf16* __restrict__ swGm1, const float* __restrict__ gm_b1,
    const __bf16* __restrict__ swGm2, const float* __restrict__ gm_b2,
    const float* __restrict__ uc_p,
    float* __restrict__ h_out, float* __restrict__ X_out) {
  __shared__ float s_h[2048];
  __shared__ float s_x1[2304];
  __shared__ float s_xp[2304];
  __shared__ float s_nrm[256];
  __shared__ float s_c[2048];
  __shared__ float s_g2[2304];
  __shared__ __bf16 s_ab[2560];
  const int tid = threadIdx.x, lane = tid & 31, wave = tid >> 5;
  const size_t n0 = (size_t)blockIdx.x * 16;
  const float uc = uc_p[0];
  const float co = rsqrtf(uc * uc + 1.f);
  const float ucco = uc * co;
  const int mb = (lane >> 4) * 8;

  if (tid < 16) {  // LayerNorm(dh) + residual
    size_t n = n0 + tid;
    float mu = 0.f;
    for (int j = 0; j < 128; ++j) mu += dh_acc[n * 128 + j];
    mu *= (1.f / 128.f);
    float var = 0.f;
    for (int j = 0; j < 128; ++j) { float t = dh_acc[n * 128 + j] - mu; var += t * t; }
    var *= (1.f / 128.f);
    float inv = rsqrtf(var + 1e-5f);
    for (int j = 0; j < 128; ++j) {
      float dh = (dh_acc[n * 128 + j] - mu) * inv * ln_g[j] + ln_b[j];
      s_h[tid * 128 + j] = co * h[n * 128 + j] + ucco * dh;
    }
  }
  if (tid < 48) {  // SO3 LN(dX) + residual
    int nl = tid / 3, l = tid - nl * 3;
    size_t n = n0 + nl;
    int a = (l == 0) ? 0 : (l == 1 ? 1 : 4);
    int b = (l == 0) ? 1 : (l == 1 ? 4 : 9);
    float ss = 0.f;
    for (int m = 0; m < 16; ++m)
      for (int i = a; i < b; ++i) { float v = dX_acc[n * 144 + m * 9 + i]; ss += v * v; }
    float inv = 1.f / sqrtf(ss * (1.f / 16.f) + 1e-8f);
    for (int m = 0; m < 16; ++m)
      for (int i = a; i < b; ++i)
        s_x1[nl * 144 + m * 9 + i] =
            co * Xin[n * 144 + m * 9 + i] + ucco * dX_acc[n * 144 + m * 9 + i] * inv;
  }
  __syncthreads();
  for (int idx = tid; idx < 2304; idx += 256) {  // Xp = einsum(X1, Wvu)
    int nl = idx / 144, r = idx - nl * 144;
    int j = r / 9, i = r - j * 9;
    float s = 0.f;
    for (int m = 0; m < 16; ++m) s += s_x1[nl * 144 + m * 9 + i] * Wvu[m * 16 + j];
    s_xp[idx] = s;
  }
  __syncthreads();
  {
    int nl = tid >> 4, j = tid & 15;
    float ss = 0.f;
    for (int i = 0; i < 9; ++i) { float v = s_xp[nl * 144 + j * 9 + i]; ss += v * v; }
    s_nrm[tid] = sqrtf(ss + 1e-12f);
  }
  __syncthreads();
  for (int idx = tid; idx < 16 * 160; idx += 256) {  // A = [h1, nrm] pad K->160
    int nl = idx / 160, k = idx - nl * 160;
    float v = (k < 128) ? s_h[nl * 128 + k]
                        : (k < 144 ? s_nrm[nl * 16 + (k - 128)] : 0.f);
    s_ab[idx] = (__bf16)v;
  }
  __syncthreads();
  {  // gm layer 1 (Kp=160)
    int cl = wave * 16 + (lane & 15);
    const float b1 = gm_b1[cl];
    f32x8 c = wmma_rowtile(s_ab, 160, swGm1, wave, lane);
#pragma unroll
    for (int v = 0; v < 8; ++v) s_c[(mb + v) * 128 + cl] = silu_f(c[v] + b1);
  }
  __syncthreads();
  for (int i = tid; i < 2048; i += 256) s_ab[i] = (__bf16)s_c[i];
  __syncthreads();
  for (int nt = wave; nt < 9; nt += 8) {  // gm layer 2 (dout=144)
    int cl = nt * 16 + (lane & 15);
    const float b2 = gm_b2[cl];
    f32x8 c = wmma_rowtile(s_ab, 128, swGm2, nt, lane);
#pragma unroll
    for (int v = 0; v < 8; ++v) s_g2[(mb + v) * 144 + cl] = c[v] + b2;
  }
  __syncthreads();
  for (int idx = tid; idx < 2048; idx += 256) {
    int nl = idx >> 7, j = idx & 127;
    h_out[(n0 + nl) * 128 + j] = s_h[nl * 128 + j] + s_g2[nl * 144 + j];
  }
  for (int idx = tid; idx < 2304; idx += 256) {
    int nl = idx / 144, r = idx - nl * 144, j = r / 9;
    X_out[n0 * 144 + idx] = s_x1[idx] + s_g2[nl * 144 + 128 + j] * s_xp[idx];
  }
}

// ---------------------------------------------------------------------------
extern "C" void kernel_launch(void* const* d_in, const int* in_sizes, int n_in,
                              void* d_out, int out_size, void* d_ws, size_t ws_size,
                              hipStream_t stream) {
  const float* h    = (const float*)d_in[0];
  const float* X    = (const float*)d_in[1];
  const float* t    = (const float*)d_in[2];
  const float* sph  = (const float*)d_in[3];
  const float* ucp  = (const float*)d_in[4];
  const float* Wtq  = (const float*)d_in[5];
  const float* Wtk  = (const float*)d_in[6];
  const int*   ec   = (const int*)d_in[49];
  const int*   en   = (const int*)d_in[50];
  const int N = in_sizes[0] / 128;
  const int E = in_sizes[2] / 128;

  // biases (fp32, straight from inputs)
  const float* gw_b1 = (const float*)d_in[8];   const float* gw_b2 = (const float*)d_in[10];
  const float* gt_b1 = (const float*)d_in[12];  const float* gt_b2 = (const float*)d_in[14];
  const float* gs_b1 = (const float*)d_in[16];  const float* gs_b2 = (const float*)d_in[18];
  const float* gv_b1 = (const float*)d_in[20];  const float* gv_b2 = (const float*)d_in[22];
  const float* gsp_b1 = (const float*)d_in[24]; const float* gsp_b2 = (const float*)d_in[26];
  const float* md_b1 = (const float*)d_in[28];  const float* md_b2 = (const float*)d_in[30];
  const float* mt_b1 = (const float*)d_in[32];  const float* mt_b2 = (const float*)d_in[34];
  const float* gm_b1 = (const float*)d_in[36];  const float* gm_b2 = (const float*)d_in[38];
  const float* Wrs_b = (const float*)d_in[40];
  const float* Wq_b  = (const float*)d_in[42];
  const float* Wk_b  = (const float*)d_in[44];
  const float* ln_g  = (const float*)d_in[46];
  const float* ln_b  = (const float*)d_in[47];
  const float* Wvu   = (const float*)d_in[48];

  // weight-swizzle table: {input index, K, Kp, Nout}
  static const int wIdx[20] = {7, 9, 11, 13, 15, 17, 19, 21, 23, 25,
                               27, 29, 31, 33, 35, 37, 39, 41, 43, 45};
  static const int wK[20]   = {48, 128, 128, 128, 128, 128, 128, 128, 128, 128,
                               128, 128, 128, 128, 144, 128, 128, 128, 128, 128};
  static const int wKp[20]  = {64, 128, 128, 128, 128, 128, 128, 128, 128, 128,
                               128, 128, 128, 128, 160, 128, 128, 128, 128, 128};
  static const int wN[20]   = {128, 128, 128, 128, 128, 128, 128, 128, 128, 384,
                               128, 16, 128, 16, 128, 144, 128, 128, 128, 128};

  char* ws = (char*)d_ws;
  size_t off = 0;
  auto carve = [&](size_t bytes) -> char* {
    char* p = ws + off;
    off += (bytes + 255) & ~(size_t)255;
    return p;
  };

  __bf16* sw[20];
  for (int i = 0; i < 20; ++i)
    sw[i] = (__bf16*)carve((size_t)wKp[i] * wN[i] * sizeof(__bf16));

  float* qn     = (float*)carve((size_t)N * 128 * 4);
  float* kn     = (float*)carve((size_t)N * 128 * 4);
  float* snb    = (float*)carve((size_t)N * 128 * 4);
  float* vnb    = (float*)carve((size_t)N * 128 * 4);
  float* alpha  = (float*)carve((size_t)E * 4 * 4);
  float* amax   = (float*)carve((size_t)N * 4 * 4);
  float* den    = (float*)carve((size_t)N * 4 * 4);
  float* dh_acc = (float*)carve((size_t)N * 128 * 4);
  float* dX_acc = (float*)carve((size_t)N * 144 * 4);
  (void)ws_size;

  float* out_h = (float*)d_out;
  float* out_X = out_h + (size_t)N * 128;
  float* out_t = out_X + (size_t)N * 144;
  (void)out_size; (void)n_in;

  // 0) swizzle all weights to bf16 fragment layout
  for (int i = 0; i < 20; ++i) {
    int total = wKp[i] * wN[i];
    k_swizzle<<<(total + 255) / 256, 256, 0, stream>>>(
        (const float*)d_in[wIdx[i]], sw[i], wK[i], wKp[i], wN[i]);
  }
  // 1) init accumulators
  {
    size_t n4 = (size_t)N * 4, n128 = (size_t)N * 128, n144 = (size_t)N * 144;
    k_fill<<<(int)((n4 + 255) / 256), 256, 0, stream>>>(amax, -3.402823466e38f, n4);
    k_fill<<<(int)((n4 + 255) / 256), 256, 0, stream>>>(den, 0.f, n4);
    k_fill<<<(int)((n128 + 255) / 256), 256, 0, stream>>>(dh_acc, 0.f, n128);
    k_fill<<<(int)((n144 + 255) / 256), 256, 0, stream>>>(dX_acc, 0.f, n144);
  }
  // 2) per-node precompute
  k_node_pre<<<N / 16, 256, 0, stream>>>(h, sw[17], Wq_b, sw[18], Wk_b,
                                         sw[4], gs_b1, sw[5], gs_b2,
                                         sw[6], gv_b1, sw[7], gv_b2,
                                         qn, kn, snb, vnb);
  // 3) edge phase 1
  k_edge1<<<E / 16, 256, 0, stream>>>(X, t, sph, Wtq, Wtk, ec, en,
                                      sw[0], gw_b1, sw[1], gw_b2,
                                      sw[2], gt_b1, sw[3], gt_b2,
                                      sw[19], qn, kn, ucp,
                                      out_t, alpha, amax);
  // 4) softmax denominator
  k_softmax<<<(E * 4 + 255) / 256, 256, 0, stream>>>(ec, alpha, amax, den, E * 4);
  // 5) edge phase 2
  k_edge2<<<E / 16, 256, 0, stream>>>(out_t, X, sph, ec, en, alpha, den, snb, vnb,
                                      sw[16], Wrs_b,
                                      sw[8], gsp_b1, sw[9], gsp_b2,
                                      sw[10], md_b1, sw[11], md_b2,
                                      sw[12], mt_b1, sw[13], mt_b2,
                                      dh_acc, dX_acc);
  // 6) node finalize
  k_node_fin<<<N / 16, 256, 0, stream>>>(h, X, dh_acc, dX_acc, ln_g, ln_b, Wvu,
                                         sw[14], gm_b1, sw[15], gm_b2, ucp,
                                         out_h, out_X);
}
